// SSIMLoss_1108101562605
// MI455X (gfx1250) — matrix operand order repaired and checked by reading, not compiled, not run on hardware
//
#include <hip/hip_runtime.h>

// ---------------------------------------------------------------------------
// SSIM-1D loss, fused single pass + deterministic reduction.
// MI455X (gfx1250, wave32) strategy:
//   * 11-tap depthwise conv == 16x26 banded matmul -> one V_WMMA_F32_16X16X32_F16
//     per conv signal (K=26 zero-padded to 32). 5 WMMAs per 16-row x 16-output tile.
//   * fp16 A/B with fp32 accumulation; inputs are ~N(0,1) so f16 quantization
//     error (~1e-3 rel) is far below the C2=9e-4 stabilizers' effect.
//   * Aligned windows: w0 multiple of 16 -> each lane does 4x float4 (64B)
//     aligned loads; halo overlap between adjacent tiles lives in 192MB L2.
//   * Memory floor ~8.4us @ 23.3TB/s; matrix work ~2us; pointwise SSIM stays
//     on VALU (~15 ops/output).
// ---------------------------------------------------------------------------

typedef __attribute__((ext_vector_type(16))) _Float16 v16h;
typedef __attribute__((ext_vector_type(8)))  float    v8f;

#define T_LEN          48000
#define N_ROWS         512            // B*C = 16*32 independent rows
#define N_RG           (N_ROWS / 16)  // 32 row-groups of 16 rows
#define N_WIN          3001           // windows per row-group, w0 = 16*(w-1)
#define N_TILES        (N_RG * N_WIN) // 96032
#define WAVES_PER_BLK  8
#define BLOCK_SIZE     (WAVES_PER_BLK * 32)
#define N_BLOCKS       (N_TILES / WAVES_PER_BLK)  // 12004 (exact)
#define C1F            1.0e-4f
#define C2F            9.0e-4f

// gaussian(11, sigma=1.5), normalized (matches reference _gauss_window)
__constant__ float c_g[11] = {
    0.00102838f, 0.00759880f, 0.03600077f, 0.10936069f, 0.21300553f,
    0.26601172f,
    0.21300553f, 0.10936069f, 0.03600077f, 0.00759880f, 0.00102838f
};

__global__ __launch_bounds__(BLOCK_SIZE)
void ssim_tile_kernel(const float* __restrict__ pred,
                      const float* __restrict__ tgt,
                      float* __restrict__ partial)
{
    const int lane = threadIdx.x & 31;
    const int wave = threadIdx.x >> 5;
    const int tile = blockIdx.x * WAVES_PER_BLK + wave;   // always < N_TILES
    const int rg   = tile / N_WIN;
    const int win  = tile - rg * N_WIN;
    const int w0   = (win - 1) * 16;          // window start, can be -16
    const int hi   = lane >> 4;               // lane half (K 0..15 vs 16..31)
    const int n    = lane & 15;               // N index == row-in-group, M index for A

    const float* prow = pred + (long long)(rg * 16 + n) * T_LEN;
    const float* trow = tgt  + (long long)(rg * 16 + n) * T_LEN;

    // ---- A fragment: constant banded Gaussian W[m,k] = g[k-m], 0<=k-m<=10 ----
    // 16-bit A 16x32 layout (contiguous K, matching B): VGPR v, elem e holds
    //   K = 16*hi + v*2 + e,  M = lane&15
    v16h a;
#pragma unroll
    for (int v = 0; v < 8; ++v) {
#pragma unroll
        for (int e = 0; e < 2; ++e) {
            int k = hi * 16 + v * 2 + e;
            int d = k - n;
            float val = (d >= 0 && d <= 10) ? c_g[d] : 0.0f;
            a[v * 2 + e] = (_Float16)val;
        }
    }

    // ---- B data: lane holds row n, K = 16*hi + j (j=0..15), X[k]=row[w0+k] ----
    const int base = w0 + 16 * hi;
    float p[16], t[16];
    const bool interior = (w0 >= 0) && (w0 + 32 <= T_LEN);  // wave-uniform
    if (interior) {
        // 64B-aligned contiguous per lane (48000 % 16 == 0, base % 16 == 0)
        const float4* p4 = (const float4*)(prow + base);
        const float4* t4 = (const float4*)(trow + base);
#pragma unroll
        for (int i = 0; i < 4; ++i) {
            float4 pv = p4[i];
            float4 tv = t4[i];
            p[4 * i + 0] = pv.x; p[4 * i + 1] = pv.y;
            p[4 * i + 2] = pv.z; p[4 * i + 3] = pv.w;
            t[4 * i + 0] = tv.x; t[4 * i + 1] = tv.y;
            t[4 * i + 2] = tv.z; t[4 * i + 3] = tv.w;
        }
    } else {
        // boundary windows (first/last): emulate the conv's zero padding
#pragma unroll
        for (int j = 0; j < 16; ++j) {
            int idx = base + j;
            bool ok = (idx >= 0) && (idx < T_LEN);
            p[j] = ok ? prow[idx] : 0.0f;
            t[j] = ok ? trow[idx] : 0.0f;
        }
    }

    v16h bp, bt;
#pragma unroll
    for (int j = 0; j < 16; ++j) {
        bp[j] = (_Float16)p[j];
        bt[j] = (_Float16)t[j];
    }
    v16h bpp = bp * bp;   // v_pk_mul_f16
    v16h btt = bt * bt;
    v16h bpt = bp * bt;

    // ---- 5 WMMAs: mu1, mu2, conv(p^2), conv(t^2), conv(p*t) (shared A) ----
    v8f zero = {};
    v8f mu1 = __builtin_amdgcn_wmma_f32_16x16x32_f16(false, a, false, bp,  (short)0, zero, false, false);
    v8f mu2 = __builtin_amdgcn_wmma_f32_16x16x32_f16(false, a, false, bt,  (short)0, zero, false, false);
    v8f sp  = __builtin_amdgcn_wmma_f32_16x16x32_f16(false, a, false, bpp, (short)0, zero, false, false);
    v8f st  = __builtin_amdgcn_wmma_f32_16x16x32_f16(false, a, false, btt, (short)0, zero, false, false);
    v8f spt = __builtin_amdgcn_wmma_f32_16x16x32_f16(false, a, false, bpt, (short)0, zero, false, false);

    // ---- pointwise SSIM + local accumulation of (1 - ssim) ----
    // C/D layout: VGPR i holds M = i + 8*hi, N = lane&15; output pos = w0+5+M
    float acc = 0.0f;
#pragma unroll
    for (int i = 0; i < 8; ++i) {
        int pos = w0 + 5 + i + 8 * hi;
        float m1  = mu1[i], m2 = mu2[i];
        float m1s = m1 * m1, m2s = m2 * m2, m12 = m1 * m2;
        float s1  = sp[i]  - m1s;
        float s2  = st[i]  - m2s;
        float s12 = spt[i] - m12;
        float num = (2.0f * m12 + C1F) * (2.0f * s12 + C2F);
        float den = (m1s + m2s + C1F) * (s1 + s2 + C2F);
        float v   = 1.0f - num / den;
        if (pos >= 0 && pos < T_LEN) acc += v;
    }

    // ---- deterministic reduction: wave (shfl) -> block (LDS) -> d_ws ----
#pragma unroll
    for (int off = 16; off > 0; off >>= 1)
        acc += __shfl_down(acc, off, 32);

    __shared__ float wsum[WAVES_PER_BLK];
    if (lane == 0) wsum[wave] = acc;
    __syncthreads();
    if (threadIdx.x == 0) {
        float s = 0.0f;
#pragma unroll
        for (int wv = 0; wv < WAVES_PER_BLK; ++wv) s += wsum[wv];
        partial[blockIdx.x] = s;
    }
}

__global__ __launch_bounds__(256)
void ssim_reduce_kernel(const float* __restrict__ partial,
                        float* __restrict__ out)
{
    __shared__ float sm[256];
    float s = 0.0f;
    for (int i = threadIdx.x; i < N_BLOCKS; i += 256)
        s += partial[i];                      // fixed per-thread order
    sm[threadIdx.x] = s;
    __syncthreads();
    for (int off = 128; off > 0; off >>= 1) {
        if (threadIdx.x < off) sm[threadIdx.x] += sm[threadIdx.x + off];
        __syncthreads();
    }
    if (threadIdx.x == 0)
        out[0] = sm[0] * (1.0f / (16.0f * 32.0f * 48000.0f));
}

extern "C" void kernel_launch(void* const* d_in, const int* in_sizes, int n_in,
                              void* d_out, int out_size, void* d_ws, size_t ws_size,
                              hipStream_t stream)
{
    const float* pred = (const float*)d_in[0];
    const float* tgt  = (const float*)d_in[1];
    float* partial = (float*)d_ws;            // N_BLOCKS floats (~48 KB)

    ssim_tile_kernel<<<N_BLOCKS, BLOCK_SIZE, 0, stream>>>(pred, tgt, partial);
    ssim_reduce_kernel<<<1, 256, 0, stream>>>(partial, (float*)d_out);
}
